// STAN_5729486372880
// MI455X (gfx1250) — matrix operand order repaired and checked by reading, not compiled, not run on hardware
//
#include <hip/hip_runtime.h>
#include <hip/hip_bf16.h>

// ---------------------------------------------------------------------------
// STAN GNN forward for MI455X (gfx1250, wave32).
//   GAT1 (4 heads x 64) -> ELU -> GAT2 (1 head x 128) -> ELU -> GRU(h0=0)
//   -> heads (pred_I/pred_R/alpha,beta) -> 15-step SIR rollout.
// Dense GEMMs: v_wmma_f32_16x16x32_f16, B pre-packed into fragment layout so
// each lane loads its 16 B halves with one contiguous 32B load (2x b128).
// Each wave computes a 16x32 C tile (2 WMMAs per K-step, A reused).
// Segment softmax over 660k edges uses native global f32 atomics.
// Workspace requirement: ~63.2 MB (floats 52.8MB + f16 staging 10.4MB).
// ---------------------------------------------------------------------------

typedef _Float16 v16h __attribute__((ext_vector_type(16)));
typedef _Float16 v8h  __attribute__((ext_vector_type(8)));
typedef float    v8f  __attribute__((ext_vector_type(8)));

#define TPB 256

// ---------- small helpers ----------
__device__ __forceinline__ void atomicMaxF(float* addr, float v) {
  // monotonic int mapping: works with -inf init; integer atomics only.
  int vi = __float_as_int(v);
  if (vi >= 0) atomicMax((int*)addr, vi);
  else         atomicMin((unsigned int*)addr, (unsigned int)vi);
}
__device__ __forceinline__ float sigm(float x) { return 1.f / (1.f + __expf(-x)); }

// ---------- utility kernels ----------
__global__ void stan_fill(float* __restrict__ p, float v, long long n) {
  long long t = (long long)blockIdx.x * blockDim.x + threadIdx.x;
  if (t < n) p[t] = v;
}

__global__ void stan_cvt_f16(const float* __restrict__ in, _Float16* __restrict__ out,
                             long long n) {
  long long t = (long long)blockIdx.x * blockDim.x + threadIdx.x;
  if (t < n) out[t] = (_Float16)in[t];
}

// Pack a KxN weight (optionally stored transposed as NxK) into WMMA B-fragment
// layout: outp[((kb*ntiles + nt)*32 + lane)*16 + i] holds B[kb*32 + (i>>3)*16 + hi*8 + (i&7)][nt*16 + l16]
// with lane = hi*16 + l16.  Each lane's fragment = 16 contiguous halves (32B).
__global__ void stan_pack_b(const float* __restrict__ in, _Float16* __restrict__ outp,
                            int K, int N, int transpose) {
  int t = blockIdx.x * blockDim.x + threadIdx.x;
  if (t >= K * N) return;
  int k = t / N, n = t - k * N;
  float v = transpose ? in[(size_t)n * K + k] : in[t];
  int kb = k >> 5, hi = (k >> 3) & 1, i = (k & 7) + (((k >> 4) & 1) << 3);
  int nt = n >> 4, l16 = n & 15;
  int lane = hi * 16 + l16;
  int ntiles = N >> 4;
  outp[(((size_t)kb * ntiles + nt) * 32 + lane) * 16 + i] = (_Float16)v;
}

// ---------- WMMA GEMM: C(MxN,f32) = A(MxK,f16 row-major) * Bpacked(KxN)
// One 16x32 C tile per wave; K is compile-time -> fully unrolled WMMA chain.
template <int K>
__global__ void stan_wmma_gemm(const _Float16* __restrict__ A,
                               const _Float16* __restrict__ Bp,
                               float* __restrict__ C,
                               int M, int N) {
  const int lane = threadIdx.x & 31;
  const int wave = threadIdx.x >> 5;
  const int tile = blockIdx.x * (blockDim.x >> 5) + wave;   // over (M/16)*(N/32)
  const int npair = N >> 5;
  const int mt = tile / npair;
  const int np = tile - mt * npair;
  if (mt * 16 >= M) return;             // whole-wave uniform exit: EXEC all-1s for WMMA
  const int hi  = lane >> 4;            // 0: lanes 0-15, 1: lanes 16-31
  const int l16 = lane & 15;
  const int ntiles = N >> 4;

  // A fragment: row m = mt*16+l16; halves 0..7 -> k = kb8..kb8+7, 8..15 -> +16
  const _Float16* Arow = A + (size_t)(mt * 16 + l16) * K + hi * 8;
  // Packed B: lane's 16 halves are contiguous; two n-tiles are 512 halves apart.
  const _Float16* B0 = Bp + ((size_t)(np * 2) * 32 + lane) * 16;
  __builtin_prefetch(B0, 0, 3);

  v8f c0 = {}, c1 = {};
#pragma unroll
  for (int kb = 0; kb < K / 32; ++kb) {
    const int k0 = kb * 32;
    v8h a0 = *(const v8h*)(Arow + k0);
    v8h a1 = *(const v8h*)(Arow + k0 + 16);
    v16h a = __builtin_shufflevector(a0, a1, 0,1,2,3,4,5,6,7,8,9,10,11,12,13,14,15);
    const _Float16* bb = B0 + (size_t)kb * ntiles * 512;
    v16h b0 = *(const v16h*)(bb);         // 32B contiguous -> 2x global_load_b128
    v16h b1 = *(const v16h*)(bb + 512);
    c0 = __builtin_amdgcn_wmma_f32_16x16x32_f16(false, a, false, b0, (short)0, c0,
                                                false, false);
    c1 = __builtin_amdgcn_wmma_f32_16x16x32_f16(false, a, false, b1, (short)0, c1,
                                                false, false);
  }
  // C/D layout: lane col n = l16; VGPR r -> M = r + 8*hi
  float* Cc = C + (size_t)(np * 32 + l16);
#pragma unroll
  for (int r = 0; r < 8; ++r) {
    const size_t row = (size_t)(mt * 16 + hi * 8 + r) * N;
    Cc[row]      = c0[r];
    Cc[row + 16] = c1[r];
  }
}

// ---------- attention scores: al/ar[n,h] = sum_c xl[n,h,c]*a_{src,dst}[h,c]
__global__ void stan_attn_scores(const float* __restrict__ xl,
                                 const float* __restrict__ a_src,
                                 const float* __restrict__ a_dst,
                                 float* __restrict__ al, float* __restrict__ ar,
                                 int n, int H, int CH) {
  int t = blockIdx.x * blockDim.x + threadIdx.x;
  if (t >= n * H) return;
  int node = t / H, h = t - node * H;
  const float* xr = xl + (size_t)node * H * CH + (size_t)h * CH;
  const float* as = a_src + h * CH;
  const float* ad = a_dst + h * CH;
  float sa = 0.f, sd = 0.f;
  for (int c = 0; c < CH; ++c) { float v = xr[c]; sa += v * as[c]; sd += v * ad[c]; }
  al[t] = sa; ar[t] = sd;
}

// ---------- edge pass A: e = leaky_relu(al[src]+ar[dst]); store; segment max
__global__ void stan_edge_a(const int* __restrict__ EI,
                            const float* __restrict__ al, const float* __restrict__ ar,
                            float* __restrict__ ebuf, float* __restrict__ mbuf,
                            int E_real, int E_total, int H) {
  int e = blockIdx.x * blockDim.x + threadIdx.x;
  if (e >= E_total) return;
  int s, d;
  if (e < E_real) { s = EI[e]; d = EI[E_real + e]; }
  else            { s = d = e - E_real; }
  for (int h = 0; h < H; ++h) {
    float v = al[s * H + h] + ar[d * H + h];
    v = v > 0.f ? v : 0.2f * v;
    ebuf[(size_t)e * H + h] = v;
    atomicMaxF(&mbuf[d * H + h], v);
  }
}

// ---------- edge pass B: ee = exp(e - m[dst]); store; segment sum
__global__ void stan_edge_b(const int* __restrict__ EI,
                            float* __restrict__ ebuf, const float* __restrict__ mbuf,
                            float* __restrict__ sbuf,
                            int E_real, int E_total, int H) {
  int e = blockIdx.x * blockDim.x + threadIdx.x;
  if (e >= E_total) return;
  int d = (e < E_real) ? EI[E_real + e] : (e - E_real);
  for (int h = 0; h < H; ++h) {
    float ee = __expf(ebuf[(size_t)e * H + h] - mbuf[d * H + h]);
    ebuf[(size_t)e * H + h] = ee;
    atomicAdd(&sbuf[d * H + h], ee);
  }
}

// ---------- edge pass C: acc[dst] += ee * xl[src]  (normalization deferred)
__global__ void stan_edge_c(const int* __restrict__ EI,
                            const float* __restrict__ ebuf,
                            const float* __restrict__ xl,
                            float* __restrict__ acc,
                            int E_real, int E_total, int H, int chShift) {
  const int CH = 1 << chShift;
  long long t = (long long)blockIdx.x * blockDim.x + threadIdx.x;
  int e = (int)(t >> chShift);
  if (e >= E_total) return;
  int c = (int)t & (CH - 1);
  int s, d;
  if (e < E_real) { s = EI[e]; d = EI[E_real + e]; }
  else            { s = d = e - E_real; }
  const int F = H << chShift;
  const float* xr = xl + (size_t)s * F;
  float* ac = acc + (size_t)d * F;
  for (int h = 0; h < H; ++h) {
    int ch = (h << chShift) + c;
    atomicAdd(&ac[ch], xr[ch] * ebuf[(size_t)e * H + h]);
  }
}

// ---------- out = ELU(acc/s + bias), emitted directly as f16 for next GEMM
__global__ void stan_elu_bias_cvt(const float* __restrict__ acc,
                                  const float* __restrict__ sbuf,
                                  const float* __restrict__ bias,
                                  _Float16* __restrict__ xh,
                                  long long total, int H, int chShift, int fShift) {
  long long t = (long long)blockIdx.x * blockDim.x + threadIdx.x;
  if (t >= total) return;
  int node = (int)(t >> fShift);
  int f = (int)t & ((1 << fShift) - 1);
  int h = f >> chShift;
  float v = acc[t] / sbuf[node * H + h] + bias[f];
  v = v > 0.f ? v : (__expf(v) - 1.f);
  xh[t] = (_Float16)v;
}

// ---------- GRU with h0 = 0:  h = (1-z)*tanh(ih + r*b_hh_n)
__global__ void stan_gru(const float* __restrict__ gi,
                         const float* __restrict__ b_ih, const float* __restrict__ b_hh,
                         float* __restrict__ h_out, int n) {
  int t = blockIdx.x * blockDim.x + threadIdx.x;
  if (t >= n * 128) return;
  int node = t >> 7, j = t & 127;
  const float* g = gi + (size_t)node * 384;
  float r  = sigm(g[j]       + b_ih[j]       + b_hh[j]);
  float z  = sigm(g[128 + j] + b_ih[128 + j] + b_hh[128 + j]);
  float nn = tanhf(g[256 + j] + b_ih[256 + j] + r * b_hh[256 + j]);
  h_out[t] = (1.f - z) * nn;
}

// ---------- pred_I / pred_R heads: hc=[h,cI,cR] (130) @ W^T + b
__global__ void stan_heads(const float* __restrict__ h,
                           const float* __restrict__ cI, const float* __restrict__ cR,
                           const float* __restrict__ W_I, const float* __restrict__ b_I,
                           const float* __restrict__ W_R, const float* __restrict__ b_R,
                           float* __restrict__ out, int n) {
  int t = blockIdx.x * blockDim.x + threadIdx.x;
  if (t >= n * 30) return;
  int node = t / 30, o = t - node * 30;
  const float* hr = h + (size_t)node * 128;
  const float* W; float b;
  if (o < 15) { W = W_I + o * 130;        b = b_I[o]; }
  else        { W = W_R + (o - 15) * 130; b = b_R[o - 15]; }
  float s = b;
  for (int k = 0; k < 128; ++k) s += hr[k] * W[k];
  s += cI[node] * W[128] + cR[node] * W[129];
  if (o < 15) out[(size_t)node * 15 + o] = s;
  else        out[(size_t)n * 15 + (size_t)node * 15 + (o - 15)] = s;
}

// ---------- alpha/beta + 15-step SIR rollout (S held fixed, per reference)
__global__ void stan_sir(const float* __restrict__ h,
                         const float* __restrict__ cI, const float* __restrict__ cR,
                         const float* __restrict__ W_sir, const float* __restrict__ b_sir,
                         const float* __restrict__ Npop,
                         const float* __restrict__ I0, const float* __restrict__ R0,
                         float* __restrict__ out, int n) {
  int node = blockIdx.x * blockDim.x + threadIdx.x;
  if (node >= n) return;
  const float* hr = h + (size_t)node * 128;
  float s0 = b_sir[0], s1 = b_sir[1];
  for (int k = 0; k < 128; ++k) { s0 += hr[k] * W_sir[k]; s1 += hr[k] * W_sir[130 + k]; }
  s0 += cI[node] * W_sir[128]       + cR[node] * W_sir[129];
  s1 += cI[node] * W_sir[130 + 128] + cR[node] * W_sir[130 + 129];
  float alpha = sigm(s0), beta = sigm(s1);
  float Nv = Npop[node];
  float I = I0[node], R = R0[node];
  float S = Nv - I - R;                        // computed once (matches reference)
  float* pI = out + (size_t)2 * n * 15 + (size_t)node * 15;
  float* pR = out + (size_t)3 * n * 15 + (size_t)node * 15;
  for (int t = 0; t < 15; ++t) {
    float dI = alpha * I * (S / Nv) - beta * I;
    float dR = beta * I;
    pI[t] = dI; pR[t] = dR;
    I += dI; R += dR;
  }
}

// ---------------------------------------------------------------------------
static inline int cdiv(long long a, int b) { return (int)((a + b - 1) / b); }

extern "C" void kernel_launch(void* const* d_in, const int* in_sizes, int n_in,
                              void* d_out, int out_size, void* d_ws, size_t ws_size,
                              hipStream_t stream) {
  const float* dynamic = (const float*)d_in[0];
  const int*   EI      = (const int*)  d_in[1];
  const float* cI      = (const float*)d_in[2];
  const float* cR      = (const float*)d_in[3];
  const float* Npop    = (const float*)d_in[4];
  const float* Iv      = (const float*)d_in[5];
  const float* Rv      = (const float*)d_in[6];
  // d_in[7] dI, d_in[8] dR: unused by reference
  const float* W1      = (const float*)d_in[9];
  const float* a_src1  = (const float*)d_in[10];
  const float* a_dst1  = (const float*)d_in[11];
  const float* b1      = (const float*)d_in[12];
  const float* W2      = (const float*)d_in[13];
  const float* a_src2  = (const float*)d_in[14];
  const float* a_dst2  = (const float*)d_in[15];
  const float* b2      = (const float*)d_in[16];
  const float* W_ih    = (const float*)d_in[17];
  // d_in[18] W_hh: dead (h0 == 0)
  const float* b_ih    = (const float*)d_in[19];
  const float* b_hh    = (const float*)d_in[20];
  const float* W_I     = (const float*)d_in[21];
  const float* b_I     = (const float*)d_in[22];
  const float* W_R     = (const float*)d_in[23];
  const float* b_R     = (const float*)d_in[24];
  const float* W_sir   = (const float*)d_in[25];
  const float* b_sir   = (const float*)d_in[26];
  float* out = (float*)d_out;

  const int N  = in_sizes[2];          // 20000 nodes
  const int E  = in_sizes[1] / 2;      // 640000 real edges
  const int ET = E + N;                // + self loops

  // ---- workspace layout (floats then halves); total 63,138,304 bytes ----
  float* wsf = (float*)d_ws;
  float* XL  = wsf;                    // 5,120,000  (xl1 / xl2)
  float* EE  = wsf + 5120000;          // 2,640,000  (edge e/ee, both layers)
  float* ACC = wsf + 7760000;          // 5,120,000  (acc1/x-scratch, acc2)
  float* AL  = wsf + 12880000;         //    80,000
  float* AR  = wsf + 12960000;         //    80,000
  float* MB  = wsf + 13040000;         //    80,000
  float* SB  = wsf + 13120000;         //    80,000
  float* GI  = wsf;                    // 7,680,000 aliases XL+EE (both dead by GRU GEMM)
  _Float16* XH = (_Float16*)(wsf + 13200000);  // 5,120,000 halves (x f16, reused 3x)
  _Float16* WH = XH + 5120000;                 //    49,152 halves (packed B, reused)

  float* Hout = out + (size_t)4 * N * 15;      // h section of output

  const float NEG_INF = -__builtin_huge_valf();

  // ================= GAT layer 1: 4 heads x 64 =================
  stan_cvt_f16<<<cdiv((long long)N * 128, TPB), TPB, 0, stream>>>(dynamic, XH, (long long)N * 128);
  stan_pack_b<<<cdiv(128 * 256, TPB), TPB, 0, stream>>>(W1, WH, 128, 256, 0);
  stan_wmma_gemm<128><<<cdiv((N / 16) * (256 / 32), 8), TPB, 0, stream>>>(XH, WH, XL, N, 256);
  stan_attn_scores<<<cdiv(N * 4, TPB), TPB, 0, stream>>>(XL, a_src1, a_dst1, AL, AR, N, 4, 64);
  stan_fill<<<cdiv(N * 4, TPB), TPB, 0, stream>>>(MB, NEG_INF, N * 4);
  stan_fill<<<cdiv(N * 4, TPB), TPB, 0, stream>>>(SB, 0.f, N * 4);
  stan_fill<<<cdiv((long long)N * 256, TPB), TPB, 0, stream>>>(ACC, 0.f, (long long)N * 256);
  stan_edge_a<<<cdiv(ET, TPB), TPB, 0, stream>>>(EI, AL, AR, EE, MB, E, ET, 4);
  stan_edge_b<<<cdiv(ET, TPB), TPB, 0, stream>>>(EI, EE, MB, SB, E, ET, 4);
  stan_edge_c<<<cdiv((long long)ET << 6, TPB), TPB, 0, stream>>>(EI, EE, XL, ACC, E, ET, 4, 6);
  stan_elu_bias_cvt<<<cdiv((long long)N * 256, TPB), TPB, 0, stream>>>(ACC, SB, b1, XH,
                                                                       (long long)N * 256, 4, 6, 8);

  // ================= GAT layer 2: 1 head x 128 =================
  stan_pack_b<<<cdiv(256 * 128, TPB), TPB, 0, stream>>>(W2, WH, 256, 128, 0);
  stan_wmma_gemm<256><<<cdiv((N / 16) * (128 / 32), 8), TPB, 0, stream>>>(XH, WH, XL, N, 128);
  stan_attn_scores<<<cdiv(N, TPB), TPB, 0, stream>>>(XL, a_src2, a_dst2, AL, AR, N, 1, 128);
  stan_fill<<<cdiv(N, TPB), TPB, 0, stream>>>(MB, NEG_INF, N);
  stan_fill<<<cdiv(N, TPB), TPB, 0, stream>>>(SB, 0.f, N);
  stan_fill<<<cdiv((long long)N * 128, TPB), TPB, 0, stream>>>(ACC, 0.f, (long long)N * 128);
  stan_edge_a<<<cdiv(ET, TPB), TPB, 0, stream>>>(EI, AL, AR, EE, MB, E, ET, 1);
  stan_edge_b<<<cdiv(ET, TPB), TPB, 0, stream>>>(EI, EE, MB, SB, E, ET, 1);
  stan_edge_c<<<cdiv((long long)ET << 7, TPB), TPB, 0, stream>>>(EI, EE, XL, ACC, E, ET, 1, 7);
  stan_elu_bias_cvt<<<cdiv((long long)N * 128, TPB), TPB, 0, stream>>>(ACC, SB, b2, XH,
                                                                       (long long)N * 128, 1, 7, 7);

  // ================= GRU (h0 = 0) =================
  stan_pack_b<<<cdiv(384 * 128, TPB), TPB, 0, stream>>>(W_ih, WH, 128, 384, 1); // B = W_ih^T
  stan_wmma_gemm<128><<<cdiv((N / 16) * (384 / 32), 8), TPB, 0, stream>>>(XH, WH, GI, N, 384);
  stan_gru<<<cdiv((long long)N * 128, TPB), TPB, 0, stream>>>(GI, b_ih, b_hh, Hout, N);

  // ================= heads + SIR rollout =================
  stan_heads<<<cdiv(N * 30, TPB), TPB, 0, stream>>>(Hout, cI, cR, W_I, b_I, W_R, b_R, out, N);
  stan_sir<<<cdiv(N, TPB), TPB, 0, stream>>>(Hout, cI, cR, W_sir, b_sir, Npop, Iv, Rv, out, N);
}
